// DynamicConv1dTBC_12266426597626
// MI455X (gfx1250) — matrix operand order repaired and checked by reading, not compiled, not run on hardware
//
#include <hip/hip_runtime.h>
#include <hip/hip_bf16.h>

// Problem constants (reference: T=2048, B=4, C=1024, H=16, K=15, PAD_L=14)
#define TT_     2048
#define BB_     4
#define CC_     1024
#define HH_     16
#define KK_     15
#define PADL_   14
#define NCOLS_  (HH_ * KK_)          // 240 real logit columns
#define NCT_    16                   // padded to 16 column tiles (tile 15 = zeros)
#define KSTEPS_ (CC_ / 32)           // 32 k-steps of 32 for 16x16x32 WMMA

typedef __attribute__((ext_vector_type(16))) __bf16 v16bf;
typedef __attribute__((ext_vector_type(8)))  __bf16 v8bf;
typedef __attribute__((ext_vector_type(4)))  __bf16 v4bf;
typedef __attribute__((ext_vector_type(8)))  float  v8f;

// LDS layout (bytes):
//   xs     : 30 x 1028 fp32  = 123360   (x window, conv + source of bf16 copy)
//   logits : 16 x 256  fp32  =  16384   (logits -> softmax weights in place)
//   xbf    : 16 x 1032 bf16  =  33024   (A-operand rows t0..t0+15, pre-converted)
#define XS_ROWS    30
#define XS_STRIDE  1028              // 1028 % 64 == 4 -> rows 0..15 distinct banks
#define LG_STRIDE  256
#define XBF_STRIDE 1032              // bf16 stride; 516 dwords % 64 == 4
#define SMEM_FLOATS (XS_ROWS * XS_STRIDE + 16 * LG_STRIDE)   // 34936 floats
#define SMEM_BYTES  (SMEM_FLOATS * 4 + 16 * XBF_STRIDE * 2)  // 172768 B < 320 KB

// ---------------------------------------------------------------------------
// Kernel 0: pre-swizzle Wlin (240 x 1024 fp32) into the per-lane B-fragment
// layout of v_wmma_f32_16x16x32_bf16, padded to 16 column tiles (tile 15 = 0):
//   swz[((ct*32 + ks)*32 + lane)*16 + j] = bf16( Wlin[ct*16 + (lane&15)]
//                                                    [ks*32 + (lane>=16?8:0) + (j&7) + (j>=8?16:0)] )
// Total 16*32*32*16 = 262144 bf16 = 512 KB in d_ws.
// ---------------------------------------------------------------------------
__global__ void dynconv_swizzle_w(const float* __restrict__ Wlin,
                                  __bf16* __restrict__ swz) {
    int e = blockIdx.x * blockDim.x + threadIdx.x;     // 0 .. 262143
    if (e >= NCT_ * KSTEPS_ * 32 * 16) return;
    int j    = e & 15;
    int lane = (e >> 4) & 31;
    int ks   = (e >> 9) & 31;
    int ct   = e >> 14;                                // 0..15
    int n = ct * 16 + (lane & 15);
    int c = ks * 32 + ((lane >= 16) ? 8 : 0) + (j & 7) + ((j >> 3) << 4);
    float v = (n < NCOLS_) ? Wlin[n * CC_ + c] : 0.0f; // zero-pad tile 15
    swz[e] = (__bf16)v;
}

// ---------------------------------------------------------------------------
// Kernel 1: fused  logits-GEMM (bf16 WMMA) -> softmax -> dynamic conv.
// One block = 16 time steps of one batch element. 256 threads = 8 waves.
// Branch-free WMMA phase: every wave owns exactly 2 column tiles.
// ---------------------------------------------------------------------------
__global__ void dynconv_fused(const float* __restrict__ x,
                              const __bf16* __restrict__ swzW,
                              float* __restrict__ out) {
    extern __shared__ float smem[];
    float*  xs     = smem;
    float*  logits = smem + XS_ROWS * XS_STRIDE;
    __bf16* xbf    = (__bf16*)(smem + SMEM_FLOATS);

    const int tid  = threadIdx.x;
    const int lane = tid & 31;
    const int wave = tid >> 5;
    const int t0   = blockIdx.x * 16;
    const int b    = blockIdx.y;

    // ---- Phase 1: stage x[t0-14 .. t0+15, b, :] into LDS (zero-fill t<0), ----
    // ----          and a bf16 copy of rows t0..t0+15 for the A operand.    ----
    #pragma unroll 1
    for (int i = tid; i < XS_ROWS * 256; i += 256) {
        int row = i >> 8;            // uniform per iteration: 0..29
        int c4  = i & 255;           // float4 index within the 1024 channels
        int tg  = t0 - PADL_ + row;
        float4 v = make_float4(0.f, 0.f, 0.f, 0.f);
        if (tg >= 0)                 // right edge never overruns: t0+15 <= T-1
            v = *reinterpret_cast<const float4*>(
                    &x[(size_t)tg * (BB_ * CC_) + (size_t)b * CC_ + c4 * 4]);
        *reinterpret_cast<float4*>(&xs[row * XS_STRIDE + c4 * 4]) = v;
        if (row >= PADL_) {          // uniform branch
            v4bf bv;
            bv[0] = (__bf16)v.x; bv[1] = (__bf16)v.y;
            bv[2] = (__bf16)v.z; bv[3] = (__bf16)v.w;
            *reinterpret_cast<v4bf*>(&xbf[(row - PADL_) * XBF_STRIDE + c4 * 4]) = bv;
        }
    }
    __syncthreads();

    // ---- Phase 2: 16x256 logit GEMM via v_wmma_f32_16x16x32_bf16 ----
    {
        const int ct0 = wave * 2;        // 0,2,...,14
        const int ct1 = ct0 + 1;         // 1,3,...,15 (15 = zero pad tile)

        const int laneRow = lane & 15;
        const int koffA   = (lane < 16) ? 0 : 8;     // A-matrix lane K offset
        const __bf16* arow = &xbf[laneRow * XBF_STRIDE];
        const v16bf* bw0 = reinterpret_cast<const v16bf*>(swzW) + (ct0 * KSTEPS_) * 32 + lane;
        const v16bf* bw1 = reinterpret_cast<const v16bf*>(swzW) + (ct1 * KSTEPS_) * 32 + lane;

        v8f acc0 = {};
        v8f acc1 = {};

        #pragma unroll 4
        for (int ks = 0; ks < KSTEPS_; ++ks) {
            // A fragment: two 16B LDS loads (runs K = koffA+0..7 and 16+koffA+0..7)
            const __bf16* p = arow + ks * 32 + koffA;
            v8bf lo = *reinterpret_cast<const v8bf*>(p);
            v8bf hi = *reinterpret_cast<const v8bf*>(p + 16);
            v16bf a = __builtin_shufflevector(lo, hi,
                        0, 1, 2, 3, 4, 5, 6, 7, 8, 9, 10, 11, 12, 13, 14, 15);

            // B fragments: one coalesced 32B/lane load each (L2-resident)
            v16bf b0 = bw0[ks * 32];
            acc0 = __builtin_amdgcn_wmma_f32_16x16x32_bf16(
                       false, a, false, b0, (short)0, acc0, false, false);
            v16bf b1 = bw1[ks * 32];
            acc1 = __builtin_amdgcn_wmma_f32_16x16x32_bf16(
                       false, a, false, b1, (short)0, acc1, false, false);
        }

        // C/D layout: VGPR r -> (m = r + (lane>=16 ? 8 : 0), n = lane&15)
        const int mbase = (lane >= 16) ? 8 : 0;
        const int nloc  = lane & 15;
        #pragma unroll
        for (int r = 0; r < 8; ++r)
            logits[(r + mbase) * LG_STRIDE + ct0 * 16 + nloc] = acc0[r];
        #pragma unroll
        for (int r = 0; r < 8; ++r)
            logits[(r + mbase) * LG_STRIDE + ct1 * 16 + nloc] = acc1[r];
    }
    __syncthreads();

    // ---- Phase 3: softmax over the 15 taps, in place. One thread per (t,h). ----
    {
        int t = tid >> 4;
        int h = tid & 15;
        float* p = &logits[t * LG_STRIDE + h * KK_];
        float v[KK_];
        float m = -1e30f;
        #pragma unroll
        for (int k = 0; k < KK_; ++k) { v[k] = p[k]; m = fmaxf(m, v[k]); }
        float s = 0.f;
        #pragma unroll
        for (int k = 0; k < KK_; ++k) { v[k] = __expf(v[k] - m); s += v[k]; }
        float inv = __frcp_rn(s);
        #pragma unroll
        for (int k = 0; k < KK_; ++k) p[k] = v[k] * inv;
    }
    __syncthreads();

    // ---- Phase 4: dynamic causal conv, fp32 from LDS, coalesced stores ----
    // out[t0+tt, b, c] = sum_k w[tt, c>>6, k] * xs[tt + k, c]
    #pragma unroll 1
    for (int tt = 0; tt < 16; ++tt) {
        float* orow = &out[(size_t)(t0 + tt) * (BB_ * CC_) + (size_t)b * CC_];
        #pragma unroll
        for (int cc = 0; cc < 4; ++cc) {
            int c = cc * 256 + tid;
            const float* swp = &logits[tt * LG_STRIDE + (c >> 6) * KK_];
            float acc = 0.f;
            #pragma unroll
            for (int k = 0; k < KK_; ++k)
                acc = fmaf(swp[k], xs[(tt + k) * XS_STRIDE + c], acc);
            orow[c] = acc;
        }
    }
}

extern "C" void kernel_launch(void* const* d_in, const int* in_sizes, int n_in,
                              void* d_out, int out_size, void* d_ws, size_t ws_size,
                              hipStream_t stream) {
    const float* x    = (const float*)d_in[0];   // (T, B, C) fp32
    const float* Wlin = (const float*)d_in[1];   // (H*K, C) fp32
    float* out        = (float*)d_out;           // (T, B, C) fp32
    __bf16* swz       = (__bf16*)d_ws;           // 512 KB swizzled bf16 weights

    // Kernel 0: swizzle weights into WMMA B-fragment layout (262144 elements)
    {
        int n = NCT_ * KSTEPS_ * 32 * 16;
        dynconv_swizzle_w<<<dim3(n / 256), dim3(256), 0, stream>>>(Wlin, swz);
    }
    // Kernel 1: fused GEMM + softmax + dynamic conv
    {
        dim3 grid(TT_ / 16, BB_);
        dim3 block(256);
        dynconv_fused<<<grid, block, SMEM_BYTES, stream>>>(x, swz, out);
    }
}
